// Model_66460323938320
// MI455X (gfx1250) — matrix operand order repaired and checked by reading, not compiled, not run on hardware
//
#include <hip/hip_runtime.h>
#include <hip/hip_bf16.h>
#include <math.h>

// ---------------------------------------------------------------------------
// Types for CDNA5 WMMA bf16
// ---------------------------------------------------------------------------
typedef __attribute__((ext_vector_type(16))) __bf16 v16bf;
typedef __attribute__((ext_vector_type(8)))  __bf16 v8bf;
typedef __attribute__((ext_vector_type(8)))  float  v8f;

union AFrag { v16bf v; v8bf h[2]; };

__device__ __forceinline__ float wave_sum(float v) {
  #pragma unroll
  for (int o = 16; o > 0; o >>= 1) v += __shfl_xor(v, o, 32);
  return v;
}
__device__ __forceinline__ float wave_max(float v) {
  #pragma unroll
  for (int o = 16; o > 0; o >>= 1) v = fmaxf(v, __shfl_xor(v, o, 32));
  return v;
}

// ---------------------------------------------------------------------------
// Constants (padded)
//   B=8, N=500 -> Np=512, Q=25 -> Qp=32, D=512, F=3372 -> Fp=3392, C=80
// ---------------------------------------------------------------------------
#define NB    8
#define NN    500
#define NP    512
#define NQ    25
#define QP    32
#define DD    512
#define FF    3372
#define FP    3392
#define NC    80
#define ROWS  (NB*NP)     // 4096
#define QROWS (NB*QP)     // 256

// ---------------------------------------------------------------------------
// Packing / conversion kernels
// ---------------------------------------------------------------------------
__global__ void k_pack_nflat(const float* __restrict__ elmo,
                             const float* __restrict__ glove,
                             __bf16* __restrict__ out) {
  int f = blockIdx.x * blockDim.x + threadIdx.x;
  int row = blockIdx.y;                       // 0..4095
  if (f >= FP) return;
  int b = row >> 9, n = row & 511;
  float v = 0.f;
  if (n < NN) {
    if (f < 3072)        v = elmo[((size_t)(b*NN + n))*3072 + f];
    else if (f < FF)     v = glove[((size_t)(b*NN + n))*300 + (f - 3072)];
  }
  out[(size_t)row * FP + f] = (__bf16)v;
}

__global__ void k_pack_qflat(const float* __restrict__ elmo,
                             const float* __restrict__ glove,
                             __bf16* __restrict__ out) {
  int f = blockIdx.x * blockDim.x + threadIdx.x;
  int row = blockIdx.y;                       // 0..255 (b*32+q)
  if (f >= FP) return;
  int b = row >> 5, q = row & 31;
  float v = 0.f;
  if (q < NQ) {
    if (f < 3072)        v = elmo[((size_t)(b*NQ + q))*3072 + f];
    else if (f < FF)     v = glove[((size_t)(b*NQ + q))*300 + (f - 3072)];
  }
  out[(size_t)row * FP + f] = (__bf16)v;
}

// W (Kact x Ncols, row-major f32)  ->  Wt (Ncols x Kp, row-major bf16)
__global__ void k_convT(const float* __restrict__ W, __bf16* __restrict__ Wt,
                        int Kact, int Ncols, int Kp) {
  int k = blockIdx.x * blockDim.x + threadIdx.x;
  int n = blockIdx.y;
  if (k >= Kp) return;
  float v = (k < Kact) ? W[(size_t)k * Ncols + n] : 0.f;
  Wt[(size_t)n * Kp + k] = (__bf16)v;
}

// adj (B,3,500,500) f32 -> adjp (B,3,512,512) bf16 zero-padded
__global__ void k_pack_adj(const float* __restrict__ adj, __bf16* __restrict__ adjp) {
  int m = blockIdx.x * blockDim.x + threadIdx.x;  // 0..511
  int rr = blockIdx.y;                            // (b*3+r)*512 + n
  if (m >= NP) return;
  int n = rr & 511, br = rr >> 9;
  float v = 0.f;
  if (n < NN && m < NN) v = adj[((size_t)br * NN + n) * NN + m];
  adjp[(size_t)rr * NP + m] = (__bf16)v;
}

// ---------------------------------------------------------------------------
// Generic bf16 WMMA GEMM: C(MxN,f32) = act(A(MxK,bf16,lda) * Bt(NxK,bf16)^T + bias)
// One wave computes a 32x64 tile (2 M-frags x 4 N-frags = 8 WMMA / K-step).
// K-loop unrolling disabled to keep live set (~130 VGPR) spill-free.
// Fused epilogues selected by MODE:
//  MODE 1: nodes_compress = tanh(.), write C; last = nc*mask -> auxF;
//          cat[:,512+col] = bf16(last)                       (cat = auxBf)
//  MODE 2: plain bias add (query_compress)
//  MODE 3: h = (.)*mask, write C; hT[b][col][n] = bf16(h)    (hT = auxBf)
//  MODE 4: att = sigmoid(.)*mask; last = att*tanh(update)+(1-att)*last
//          (update=auxIn, last=auxF rw); cat[:,512+col]=bf16(last)
//  MODE 5: C = tanh(.)  (output head)
// ---------------------------------------------------------------------------
template <int MODE>
__global__ __launch_bounds__(256, 2)
void gemm_wmma(const __bf16* __restrict__ A, int lda,
               const __bf16* __restrict__ Bt, int K,
               const float* __restrict__ bias,
               float* __restrict__ C, int ldc,
               int Mtiles32, int Ntiles64,
               const int* __restrict__ nlen,
               float* __restrict__ auxF,
               const float* __restrict__ auxIn,
               __bf16* __restrict__ auxBf) {
  int gid  = blockIdx.x * blockDim.x + threadIdx.x;
  int wave = gid >> 5;
  int lane = threadIdx.x & 31;
  if (wave >= Mtiles32 * Ntiles64) return;
  int tm = wave / Ntiles64, tn = wave % Ntiles64;

  int lo    = (lane < 16);
  int abase = lo ? 0 : 16;
  const __bf16* Arow0 = A + (size_t)(tm * 32 + (lane & 15)) * lda;
  const __bf16* Arow1 = Arow0 + (size_t)16 * lda;

  const __bf16* B0 = Bt + (size_t)(tn*64 +  0 + (lane & 15)) * K + (lo ? 0 : 16);
  const __bf16* B1 = Bt + (size_t)(tn*64 + 16 + (lane & 15)) * K + (lo ? 0 : 16);
  const __bf16* B2 = Bt + (size_t)(tn*64 + 32 + (lane & 15)) * K + (lo ? 0 : 16);
  const __bf16* B3 = Bt + (size_t)(tn*64 + 48 + (lane & 15)) * K + (lo ? 0 : 16);

  v8f acc[2][4];
  #pragma unroll
  for (int mi = 0; mi < 2; ++mi)
    #pragma unroll
    for (int j = 0; j < 4; ++j) acc[mi][j] = (v8f){};

  #pragma clang loop unroll(disable)
  for (int kk = 0; kk < K; kk += 32) {
    AFrag a0, a1;
    a0.h[0] = *(const v8bf*)(Arow0 + kk + abase);
    a0.h[1] = *(const v8bf*)(Arow0 + kk + abase + 8);
    a1.h[0] = *(const v8bf*)(Arow1 + kk + abase);
    a1.h[1] = *(const v8bf*)(Arow1 + kk + abase + 8);
    v16bf b0 = *(const v16bf*)(B0 + kk);
    v16bf b1 = *(const v16bf*)(B1 + kk);
    v16bf b2 = *(const v16bf*)(B2 + kk);
    v16bf b3 = *(const v16bf*)(B3 + kk);
    acc[0][0] = __builtin_amdgcn_wmma_f32_16x16x32_bf16(false, a0.v, false, b0, (short)0, acc[0][0], false, false);
    acc[0][1] = __builtin_amdgcn_wmma_f32_16x16x32_bf16(false, a0.v, false, b1, (short)0, acc[0][1], false, false);
    acc[0][2] = __builtin_amdgcn_wmma_f32_16x16x32_bf16(false, a0.v, false, b2, (short)0, acc[0][2], false, false);
    acc[0][3] = __builtin_amdgcn_wmma_f32_16x16x32_bf16(false, a0.v, false, b3, (short)0, acc[0][3], false, false);
    acc[1][0] = __builtin_amdgcn_wmma_f32_16x16x32_bf16(false, a1.v, false, b0, (short)0, acc[1][0], false, false);
    acc[1][1] = __builtin_amdgcn_wmma_f32_16x16x32_bf16(false, a1.v, false, b1, (short)0, acc[1][1], false, false);
    acc[1][2] = __builtin_amdgcn_wmma_f32_16x16x32_bf16(false, a1.v, false, b2, (short)0, acc[1][2], false, false);
    acc[1][3] = __builtin_amdgcn_wmma_f32_16x16x32_bf16(false, a1.v, false, b3, (short)0, acc[1][3], false, false);
  }

  int col0 = tn * 64 + (lane & 15);
  #pragma unroll
  for (int mi = 0; mi < 2; ++mi) {
    int rowLo = tm * 32 + mi * 16 + ((lane >= 16) ? 8 : 0);
    #pragma unroll
    for (int j = 0; j < 4; ++j) {
      int col = col0 + j * 16;
      float bv = bias ? bias[col] : 0.f;
      #pragma unroll
      for (int v = 0; v < 8; ++v) {
        int row = rowLo + v;
        float val = acc[mi][j][v] + bv;
        if (MODE == 1) {
          float nc = tanhf(val);
          C[(size_t)row * ldc + col] = nc;
          int bb = row >> 9, nn = row & 511;
          float mk = (nn < nlen[bb]) ? 1.f : 0.f;
          float lv = nc * mk;
          auxF[(size_t)row * DD + col] = lv;
          auxBf[(size_t)row * 1024 + 512 + col] = (__bf16)lv;
        } else if (MODE == 2) {
          C[(size_t)row * ldc + col] = val;
        } else if (MODE == 3) {
          int bb = row >> 9, nn = row & 511;
          float mk = (nn < nlen[bb]) ? 1.f : 0.f;
          float hv = val * mk;
          C[(size_t)row * ldc + col] = hv;
          auxBf[((size_t)bb * NP + col) * NP + nn] = (__bf16)hv;  // hT[b][d][n]
        } else if (MODE == 4) {
          int bb = row >> 9, nn = row & 511;
          float mk = (nn < nlen[bb]) ? 1.f : 0.f;
          float at = mk / (1.f + expf(-val));
          float up = auxIn[(size_t)row * DD + col];
          float ls = auxF[(size_t)row * DD + col];
          float nl = at * tanhf(up) + (1.f - at) * ls;
          auxF[(size_t)row * DD + col] = nl;
          auxBf[(size_t)row * 1024 + 512 + col] = (__bf16)nl;
        } else if (MODE == 5) {
          C[(size_t)row * ldc + col] = tanhf(val);
        }
      }
    }
  }
}

// update[b,n,d] = sum_r adj[b,r,n,:] @ h[b,:,d] + h[b,n,d]; also cat[:,col]=bf16
__global__ __launch_bounds__(256, 2)
void gemm_update_wmma(const __bf16* __restrict__ adjp,   // (B,3,512,512)
                      const __bf16* __restrict__ hT,     // (B,512d,512m)
                      const float*  __restrict__ h,      // (4096,512)
                      float* __restrict__ update,        // (4096,512)
                      __bf16* __restrict__ cat) {        // (4096,1024)
  int gid  = blockIdx.x * blockDim.x + threadIdx.x;
  int wave = gid >> 5;
  int lane = threadIdx.x & 31;
  if (wave >= NB * 16 * 8) return;
  int b = wave >> 7;          // / 128
  int t = wave & 127;
  int tm = t >> 3, tn = t & 7;   // tm: 32-row tiles (16), tn: 64-col tiles (8)

  int lo    = (lane < 16);
  int abase = lo ? 0 : 16;

  const __bf16* hTb = hT + (size_t)b * NP * NP;
  const __bf16* B0 = hTb + (size_t)(tn*64 +  0 + (lane & 15)) * NP + (lo ? 0 : 16);
  const __bf16* B1 = hTb + (size_t)(tn*64 + 16 + (lane & 15)) * NP + (lo ? 0 : 16);
  const __bf16* B2 = hTb + (size_t)(tn*64 + 32 + (lane & 15)) * NP + (lo ? 0 : 16);
  const __bf16* B3 = hTb + (size_t)(tn*64 + 48 + (lane & 15)) * NP + (lo ? 0 : 16);

  v8f acc[2][4];
  #pragma unroll
  for (int mi = 0; mi < 2; ++mi)
    #pragma unroll
    for (int j = 0; j < 4; ++j) acc[mi][j] = (v8f){};

  for (int r = 0; r < 3; ++r) {
    const __bf16* Arow0 = adjp + (((size_t)b * 3 + r) * NP + tm * 32 + (lane & 15)) * NP;
    const __bf16* Arow1 = Arow0 + (size_t)16 * NP;
    #pragma clang loop unroll(disable)
    for (int kk = 0; kk < NP; kk += 32) {
      AFrag a0, a1;
      a0.h[0] = *(const v8bf*)(Arow0 + kk + abase);
      a0.h[1] = *(const v8bf*)(Arow0 + kk + abase + 8);
      a1.h[0] = *(const v8bf*)(Arow1 + kk + abase);
      a1.h[1] = *(const v8bf*)(Arow1 + kk + abase + 8);
      v16bf b0 = *(const v16bf*)(B0 + kk);
      v16bf b1 = *(const v16bf*)(B1 + kk);
      v16bf b2 = *(const v16bf*)(B2 + kk);
      v16bf b3 = *(const v16bf*)(B3 + kk);
      acc[0][0] = __builtin_amdgcn_wmma_f32_16x16x32_bf16(false, a0.v, false, b0, (short)0, acc[0][0], false, false);
      acc[0][1] = __builtin_amdgcn_wmma_f32_16x16x32_bf16(false, a0.v, false, b1, (short)0, acc[0][1], false, false);
      acc[0][2] = __builtin_amdgcn_wmma_f32_16x16x32_bf16(false, a0.v, false, b2, (short)0, acc[0][2], false, false);
      acc[0][3] = __builtin_amdgcn_wmma_f32_16x16x32_bf16(false, a0.v, false, b3, (short)0, acc[0][3], false, false);
      acc[1][0] = __builtin_amdgcn_wmma_f32_16x16x32_bf16(false, a1.v, false, b0, (short)0, acc[1][0], false, false);
      acc[1][1] = __builtin_amdgcn_wmma_f32_16x16x32_bf16(false, a1.v, false, b1, (short)0, acc[1][1], false, false);
      acc[1][2] = __builtin_amdgcn_wmma_f32_16x16x32_bf16(false, a1.v, false, b2, (short)0, acc[1][2], false, false);
      acc[1][3] = __builtin_amdgcn_wmma_f32_16x16x32_bf16(false, a1.v, false, b3, (short)0, acc[1][3], false, false);
    }
  }

  int col0 = tn * 64 + (lane & 15);
  #pragma unroll
  for (int mi = 0; mi < 2; ++mi) {
    int rowLo = b * NP + tm * 32 + mi * 16 + ((lane >= 16) ? 8 : 0);
    #pragma unroll
    for (int j = 0; j < 4; ++j) {
      int col = col0 + j * 16;
      #pragma unroll
      for (int v = 0; v < 8; ++v) {
        int row = rowLo + v;
        float val = acc[mi][j][v] + h[(size_t)row * DD + col];
        update[(size_t)row * DD + col] = val;
        cat[(size_t)row * 1024 + col]  = (__bf16)val;
      }
    }
  }
}

// ---------------------------------------------------------------------------
// Attention / head kernels (small FLOPs, VALU)
// ---------------------------------------------------------------------------
__global__ void k_qw2(const float* __restrict__ qc, const float* __restrict__ W_att,
                      float* __restrict__ qw2) {
  int i = threadIdx.x;
  if (i >= QROWS) return;
  int q = i & 31;
  float acc = 0.f;
  if (q < NQ) {
    const float* r = qc + (size_t)i * DD;
    for (int d = 0; d < DD; ++d) acc += r[d] * W_att[512 + d];
  }
  qw2[i] = acc;
}

// one wave per (b,n): sim row, softmax over q, nodes2query, max-sim
__global__ void k_sim(const float* __restrict__ last, const float* __restrict__ qc,
                      const float* __restrict__ W_att, const float* __restrict__ qw2,
                      float* __restrict__ n2q, float* __restrict__ simmax) {
  int gw   = (blockIdx.x * blockDim.x + threadIdx.x) >> 5;
  int lane = threadIdx.x & 31;
  if (gw >= NB * NN) return;
  int b = gw / NN, n = gw % NN;

  const float* lrow = last + ((size_t)b * NP + n) * DD;
  float l[16], lw3[16];
  #pragma unroll
  for (int i = 0; i < 16; ++i) {
    int d = lane + 32 * i;
    l[i]   = lrow[d];
    lw3[i] = l[i] * W_att[1024 + d];
  }
  float s1p = 0.f;
  #pragma unroll
  for (int i = 0; i < 16; ++i) s1p += l[i] * W_att[lane + 32 * i];
  float s1 = wave_sum(s1p);

  float sim[NQ];
  float mx = -INFINITY;
  for (int q = 0; q < NQ; ++q) {
    const float* qrow = qc + ((size_t)b * QP + q) * DD;
    float p = 0.f;
    #pragma unroll
    for (int i = 0; i < 16; ++i) p += lw3[i] * qrow[lane + 32 * i];
    p = wave_sum(p);
    float s = s1 + qw2[b * QP + q] + p;
    sim[q] = s;
    mx = fmaxf(mx, s);
  }
  float sum = 0.f;
  for (int q = 0; q < NQ; ++q) { float e = expf(sim[q] - mx); sim[q] = e; sum += e; }
  float inv = 1.f / sum;

  float acc[16];
  #pragma unroll
  for (int i = 0; i < 16; ++i) acc[i] = 0.f;
  for (int q = 0; q < NQ; ++q) {
    float pq = sim[q] * inv;
    const float* qrow = qc + ((size_t)b * QP + q) * DD;
    #pragma unroll
    for (int i = 0; i < 16; ++i) acc[i] += pq * qrow[lane + 32 * i];
  }
  float* orow = n2q + ((size_t)b * NP + n) * DD;
  #pragma unroll
  for (int i = 0; i < 16; ++i) orow[lane + 32 * i] = acc[i];
  if (lane == 0) simmax[b * NP + n] = mx;
}

__global__ void k_bw_softmax(const float* __restrict__ simmax, float* __restrict__ p_bw) {
  int b = blockIdx.x, t = threadIdx.x;
  __shared__ float red[256];
  float m = -INFINITY;
  for (int n = t; n < NN; n += 256) m = fmaxf(m, simmax[b * NP + n]);
  red[t] = m; __syncthreads();
  for (int s = 128; s > 0; s >>= 1) { if (t < s) red[t] = fmaxf(red[t], red[t + s]); __syncthreads(); }
  float mx = red[0]; __syncthreads();
  float sm = 0.f;
  for (int n = t; n < NN; n += 256) sm += expf(simmax[b * NP + n] - mx);
  red[t] = sm; __syncthreads();
  for (int s = 128; s > 0; s >>= 1) { if (t < s) red[t] += red[t + s]; __syncthreads(); }
  float inv = 1.f / red[0];
  for (int n = t; n < NN; n += 256) p_bw[b * NP + n] = expf(simmax[b * NP + n] - mx) * inv;
}

__global__ void k_q2n(const float* __restrict__ p_bw, const float* __restrict__ nc,
                      float* __restrict__ q2n) {
  int b = blockIdx.y;
  int d = blockIdx.x * blockDim.x + threadIdx.x;
  if (d >= DD) return;
  float acc = 0.f;
  for (int n = 0; n < NN; ++n) acc += p_bw[b * NP + n] * nc[((size_t)(b * NP + n)) * DD + d];
  q2n[b * DD + d] = acc;
}

__global__ void k_build_g(const float* __restrict__ nc, const float* __restrict__ n2q,
                          const float* __restrict__ q2n, __bf16* __restrict__ g) {
  int c   = blockIdx.x * blockDim.x + threadIdx.x;   // 0..2047
  int row = blockIdx.y;                              // 0..4095
  if (c >= 2048) return;
  int b = row >> 9, n = row & 511;
  float v = 0.f;
  if (n < NN) {
    if (c < 512)        v = nc[(size_t)row * DD + c];
    else if (c < 1024)  v = n2q[(size_t)row * DD + (c - 512)];
    else if (c < 1536)  v = nc[(size_t)row * DD + (c - 1024)] * n2q[(size_t)row * DD + (c - 1024)];
    else                v = nc[(size_t)row * DD + (c - 1536)] * q2n[b * DD + (c - 1536)];
  }
  g[(size_t)row * 2048 + c] = (__bf16)v;
}

// raw[row] = tanhO[row,:] . W_o2 + b_o2 ; one wave per row
__global__ void k_raw(const float* __restrict__ tanhO, const float* __restrict__ W_o2,
                      const float* __restrict__ b_o2, float* __restrict__ raw) {
  int gw   = (blockIdx.x * blockDim.x + threadIdx.x) >> 5;
  int lane = threadIdx.x & 31;
  if (gw >= ROWS) return;
  const float* r = tanhO + (size_t)gw * 128;
  float acc = 0.f;
  #pragma unroll
  for (int k = 0; k < 4; ++k) acc += r[lane + 32 * k] * W_o2[lane + 32 * k];
  acc = wave_sum(acc);
  if (lane == 0) raw[gw] = acc + b_o2[0];
}

// out[b,c] = max_n ( bmask*raw, with exact-zero -> -inf ) ; one wave per (b,c)
__global__ void k_out(const float* __restrict__ bmask, const float* __restrict__ raw,
                      float* __restrict__ out) {
  int gw   = (blockIdx.x * blockDim.x + threadIdx.x) >> 5;
  int lane = threadIdx.x & 31;
  if (gw >= NB * NC) return;
  int b = gw / NC, c = gw % NC;
  float m = -INFINITY;
  for (int n = lane; n < NN; n += 32) {
    float v = bmask[((size_t)b * NC + c) * NN + n] * raw[b * NP + n];
    if (v == 0.f) v = -INFINITY;
    m = fmaxf(m, v);
  }
  m = wave_max(m);
  if (lane == 0) out[b * NC + c] = m;
}

// ---------------------------------------------------------------------------
// Host-side orchestration
// ---------------------------------------------------------------------------
extern "C" void kernel_launch(void* const* d_in, const int* in_sizes, int n_in,
                              void* d_out, int out_size, void* d_ws, size_t ws_size,
                              hipStream_t stream) {
  (void)in_sizes; (void)n_in; (void)out_size; (void)ws_size;

  const float* nodes_elmo  = (const float*)d_in[1];
  const float* query_elmo  = (const float*)d_in[2];
  const float* nodes_glove = (const float*)d_in[3];
  const float* query_glove = (const float*)d_in[4];
  const int*   nodes_len   = (const int*)  d_in[5];
  const float* adj         = (const float*)d_in[6];
  const float* bmask       = (const float*)d_in[7];
  const float* W_q  = (const float*)d_in[8];
  const float* b_q  = (const float*)d_in[9];
  const float* W_n  = (const float*)d_in[10];
  const float* b_n  = (const float*)d_in[11];
  const float* W_h  = (const float*)d_in[12];
  const float* b_h  = (const float*)d_in[13];
  const float* W_c  = (const float*)d_in[14];
  const float* b_c  = (const float*)d_in[15];
  const float* W_att= (const float*)d_in[16];
  const float* W_o1 = (const float*)d_in[17];
  const float* b_o1 = (const float*)d_in[18];
  const float* W_o2 = (const float*)d_in[19];
  const float* b_o2 = (const float*)d_in[20];
  float* out = (float*)d_out;

  char* w = (char*)d_ws;
  auto alloc = [&](size_t bytes) -> char* {
    char* p = w; w += (bytes + 255) & ~(size_t)255; return p;
  };
  __bf16* nflat = (__bf16*)alloc((size_t)ROWS  * FP * 2);
  __bf16* qflat = (__bf16*)alloc((size_t)QROWS * FP * 2);
  __bf16* WnT   = (__bf16*)alloc((size_t)DD * FP * 2);
  __bf16* WqT   = (__bf16*)alloc((size_t)DD * FP * 2);
  __bf16* WhT   = (__bf16*)alloc((size_t)DD * DD * 2);
  __bf16* WcT   = (__bf16*)alloc((size_t)DD * 1024 * 2);
  __bf16* Wo1T  = (__bf16*)alloc((size_t)128 * 2048 * 2);
  __bf16* adjp  = (__bf16*)alloc((size_t)NB * 3 * NP * NP * 2);
  float*  nc    = (float*) alloc((size_t)ROWS * DD * 4);
  float*  qc    = (float*) alloc((size_t)QROWS * DD * 4);
  float*  last  = (float*) alloc((size_t)ROWS * DD * 4);
  float*  hbuf  = (float*) alloc((size_t)ROWS * DD * 4);
  __bf16* hT    = (__bf16*)alloc((size_t)NB * NP * NP * 2);
  float*  upd   = (float*) alloc((size_t)ROWS * DD * 4);
  __bf16* cat   = (__bf16*)alloc((size_t)ROWS * 1024 * 2);
  float*  qw2   = (float*) alloc((size_t)QROWS * 4);
  float*  smax  = (float*) alloc((size_t)NB * NP * 4);
  float*  n2q   = (float*) alloc((size_t)ROWS * DD * 4);
  float*  p_bw  = (float*) alloc((size_t)NB * NP * 4);
  float*  q2n   = (float*) alloc((size_t)NB * DD * 4);
  __bf16* gbuf  = (__bf16*)alloc((size_t)ROWS * 2048 * 2);
  float*  tanhO = (float*) alloc((size_t)ROWS * 128 * 4);
  float*  raw   = (float*) alloc((size_t)ROWS * 4);

  const int TPB = 256;
  auto gemm_grid = [&](int mt, int nt) { return dim3((mt * nt * 32 + TPB - 1) / TPB); };

  // ---- pack / convert ----
  k_pack_nflat<<<dim3((FP + TPB - 1) / TPB, ROWS),  TPB, 0, stream>>>(nodes_elmo, nodes_glove, nflat);
  k_pack_qflat<<<dim3((FP + TPB - 1) / TPB, QROWS), TPB, 0, stream>>>(query_elmo, query_glove, qflat);
  k_convT<<<dim3((FP + TPB - 1) / TPB, DD),   TPB, 0, stream>>>(W_n,  WnT,  FF,   DD,  FP);
  k_convT<<<dim3((FP + TPB - 1) / TPB, DD),   TPB, 0, stream>>>(W_q,  WqT,  FF,   DD,  FP);
  k_convT<<<dim3((DD + TPB - 1) / TPB, DD),   TPB, 0, stream>>>(W_h,  WhT,  DD,   DD,  DD);
  k_convT<<<dim3((1024 + TPB - 1) / TPB, DD), TPB, 0, stream>>>(W_c,  WcT,  1024, DD,  1024);
  k_convT<<<dim3((2048 + TPB - 1) / TPB, 128),TPB, 0, stream>>>(W_o1, Wo1T, 2048, 128, 2048);
  k_pack_adj<<<dim3(2, NB * 3 * NP), TPB, 0, stream>>>(adj, adjp);

  // ---- feature compression GEMMs ----
  // nodes_compress + last + cat[:,512:]
  gemm_wmma<1><<<gemm_grid(ROWS / 32, DD / 64), TPB, 0, stream>>>(
      nflat, FP, WnT, FP, b_n, nc, DD, ROWS / 32, DD / 64, nodes_len, last, nullptr, cat);
  // query_compress
  gemm_wmma<2><<<gemm_grid(QROWS / 32, DD / 64), TPB, 0, stream>>>(
      qflat, FP, WqT, FP, b_q, qc, DD, QROWS / 32, DD / 64, nullptr, nullptr, nullptr, nullptr);

  // ---- 5 hops ----
  for (int hop = 0; hop < 5; ++hop) {
    // h = (last @ W_h + b_h) * mask ; also hT (bf16, transposed per batch)
    gemm_wmma<3><<<gemm_grid(ROWS / 32, DD / 64), TPB, 0, stream>>>(
        cat + 512, 1024, WhT, DD, b_h, hbuf, DD, ROWS / 32, DD / 64,
        nodes_len, nullptr, nullptr, hT);
    // update = sum_r adj@h + h ; cat[:,0:512] = bf16(update)
    gemm_update_wmma<<<gemm_grid(NB * 16, 8), TPB, 0, stream>>>(adjp, hT, hbuf, upd, cat);
    // att gate + last update ; cat[:,512:] = bf16(last)
    gemm_wmma<4><<<gemm_grid(ROWS / 32, DD / 64), TPB, 0, stream>>>(
        cat, 1024, WcT, 1024, b_c, nullptr, DD, ROWS / 32, DD / 64,
        nodes_len, last, upd, cat);
  }

  // ---- attention head ----
  k_qw2<<<1, TPB, 0, stream>>>(qc, W_att, qw2);
  k_sim<<<(NB * NN * 32 + TPB - 1) / TPB, TPB, 0, stream>>>(last, qc, W_att, qw2, n2q, smax);
  k_bw_softmax<<<NB, TPB, 0, stream>>>(smax, p_bw);
  k_q2n<<<dim3((DD + TPB - 1) / TPB, NB), TPB, 0, stream>>>(p_bw, nc, q2n);
  k_build_g<<<dim3((2048 + TPB - 1) / TPB, ROWS), TPB, 0, stream>>>(nc, n2q, q2n, gbuf);

  // ---- output head ----
  gemm_wmma<5><<<gemm_grid(ROWS / 32, 128 / 64), TPB, 0, stream>>>(
      gbuf, 2048, Wo1T, 2048, b_o1, tanhO, 128, ROWS / 32, 128 / 64,
      nullptr, nullptr, nullptr, nullptr);
  k_raw<<<(ROWS * 32 + TPB - 1) / TPB, TPB, 0, stream>>>(tanhO, W_o2, b_o2, raw);
  k_out<<<(NB * NC * 32 + TPB - 1) / TPB, TPB, 0, stream>>>(bmask, raw, out);
}
